// stMHCG_40716289966258
// MI455X (gfx1250) — matrix operand order repaired and checked by reading, not compiled, not run on hardware
//
#include <hip/hip_runtime.h>
#include <hip/hip_bf16.h>
#include <math.h>

#define NN 20000
#define NFEAT 3000
#define H1 512
#define H2 64
#define KCL 10
#define EPS 1e-5f

#define BM 128
#define BN 64
#define BK 32
#define LDP 40   // LDS row stride in bf16 (80 B = 20 banks -> conflict-free)

typedef __attribute__((ext_vector_type(16))) __bf16 v16bf;
typedef __attribute__((ext_vector_type(8)))  __bf16 bf16x8;
typedef __attribute__((ext_vector_type(4)))  __bf16 bf16x4;
typedef __attribute__((ext_vector_type(8)))  float  v8f;

enum { ACT_NONE = 0, ACT_BIAS = 1, ACT_SIGMOID = 2, ACT_SOFTPLUS_CLIP = 3, ACT_EXP_CLIP = 4 };

__device__ __forceinline__ float apply_mode(float v, float b, int mode) {
    v += b;
    switch (mode) {
        case ACT_SIGMOID:
            return 1.0f / (1.0f + expf(-v));
        case ACT_SOFTPLUS_CLIP: {
            float sp = (v > 20.0f) ? v : log1pf(expf(v));
            return fminf(fmaxf(sp, 1e-4f), 1e4f);
        }
        case ACT_EXP_CLIP:
            return fminf(fmaxf(expf(v), 1e-5f), 1e6f);
        default:
            return v;
    }
}

__device__ __forceinline__ v16bf frag16(const __bf16* p0, const __bf16* p1) {
    bf16x8 lo = *(const bf16x8*)p0;
    bf16x8 hi = *(const bf16x8*)p1;
    return __builtin_shufflevector(lo, hi, 0, 1, 2, 3, 4, 5, 6, 7,
                                           8, 9, 10, 11, 12, 13, 14, 15);
}

// ---------------------------------------------------------------------------
// Generic WMMA GEMM: C[M,Nc] = act(A[M,K] @ B[K,Nc] + bias), fp32 in/out,
// v_wmma_f32_16x16x32_bf16 with fp32 accumulate.
// Block: 256 threads = 8 waves; tile 128(M) x 64(N); BK = 32.
// Wave grid 4(M) x 2(N): each wave owns 32x32 (4 accs, 2 A frags x 2 B frags).
// Interior blocks stage tiles with unguarded float4 loads (global_load_b128).
// ---------------------------------------------------------------------------
__global__ void __launch_bounds__(256)
wmma_gemm_kernel(const float* __restrict__ A, const float* __restrict__ B,
                 const float* __restrict__ bias, float* __restrict__ C,
                 int M, int Kdim, int Ncols, int mode) {
    __shared__ __bf16 As[BM][LDP];   // [m][k]
    __shared__ __bf16 Bs[BN][LDP];   // [n][k]  (transposed for contiguous frags)

    const int tid  = threadIdx.x;
    const int lane = tid & 31;
    const int wid  = tid >> 5;
    const int wm   = wid >> 1;       // 0..3 -> 32-row slab
    const int wn   = wid & 1;        // 0..1 -> 32-col slab
    const int n0   = blockIdx.x * BN;
    const int m0   = blockIdx.y * BM;
    const int mrow  = lane & 15;
    const int khalf = lane >> 4;

    const bool a_int = (m0 + BM <= M);
    const bool b_int = (n0 + BN <= Ncols);

    v8f c00 = {}, c01 = {}, c10 = {}, c11 = {};

    for (int k0 = 0; k0 < Kdim; k0 += BK) {
        const bool kfull = (k0 + BK <= Kdim);
        __syncthreads();
        // ---- stage A [BM, BK] fp32 -> bf16
        if (a_int && kfull) {
            #pragma unroll
            for (int t = 0; t < 4; ++t) {
                int i4 = tid + t * 256;           // 1024 float4s
                int r = i4 >> 3, c4 = i4 & 7;
                float4 v = *(const float4*)(A + (size_t)(m0 + r) * Kdim + k0 + c4 * 4);
                bf16x4 h = { (__bf16)v.x, (__bf16)v.y, (__bf16)v.z, (__bf16)v.w };
                *(bf16x4*)&As[r][c4 * 4] = h;
            }
        } else {
            for (int i = tid; i < BM * BK; i += 256) {
                int r = i >> 5, c = i & 31;
                int gm = m0 + r, gk = k0 + c;
                float v = (gm < M && gk < Kdim) ? A[(size_t)gm * Kdim + gk] : 0.0f;
                As[r][c] = (__bf16)v;
            }
        }
        // ---- stage B [BK, BN] fp32 -> bf16 (store n-major)
        if (b_int && kfull) {
            #pragma unroll
            for (int t = 0; t < 2; ++t) {
                int i4 = tid + t * 256;           // 512 float4s
                int r = i4 >> 4, c4 = i4 & 15;
                float4 v = *(const float4*)(B + (size_t)(k0 + r) * Ncols + n0 + c4 * 4);
                Bs[c4 * 4 + 0][r] = (__bf16)v.x;
                Bs[c4 * 4 + 1][r] = (__bf16)v.y;
                Bs[c4 * 4 + 2][r] = (__bf16)v.z;
                Bs[c4 * 4 + 3][r] = (__bf16)v.w;
            }
        } else {
            for (int i = tid; i < BK * BN; i += 256) {
                int r = i >> 6, c = i & 63;
                int gk = k0 + r, gn = n0 + c;
                float v = (gk < Kdim && gn < Ncols) ? B[(size_t)gk * Ncols + gn] : 0.0f;
                Bs[c][r] = (__bf16)v;
            }
        }
        __syncthreads();

        // ---- fragments (ISA 7.12.2): A and B both in table k order (8h+s, 16+8h+s-8)
        const int ma0 = wm * 32 + mrow;
        v16bf a0 = frag16(&As[ma0][khalf * 8],      &As[ma0][16 + khalf * 8]);
        v16bf a1 = frag16(&As[ma0 + 16][khalf * 8], &As[ma0 + 16][16 + khalf * 8]);
        const int nb0 = wn * 32 + mrow;
        v16bf b0 = frag16(&Bs[nb0][khalf * 8],      &Bs[nb0][16 + khalf * 8]);
        v16bf b1 = frag16(&Bs[nb0 + 16][khalf * 8], &Bs[nb0 + 16][16 + khalf * 8]);

        c00 = __builtin_amdgcn_wmma_f32_16x16x32_bf16(false, a0, false, b0, (short)0, c00, false, false);
        c01 = __builtin_amdgcn_wmma_f32_16x16x32_bf16(false, a0, false, b1, (short)0, c01, false, false);
        c10 = __builtin_amdgcn_wmma_f32_16x16x32_bf16(false, a1, false, b0, (short)0, c10, false, false);
        c11 = __builtin_amdgcn_wmma_f32_16x16x32_bf16(false, a1, false, b1, (short)0, c11, false, false);
    }

    // ---- epilogue: lane n = lane&15; VGPR r -> m = r + 8*(lane>>4)
    #pragma unroll
    for (int bsel = 0; bsel < 2; ++bsel) {
        int nn = n0 + wn * 32 + bsel * 16 + mrow;
        if (nn >= Ncols) continue;
        float bv = (mode != ACT_NONE && bias != nullptr) ? bias[nn] : 0.0f;
        #pragma unroll
        for (int r = 0; r < 8; ++r) {
            int ma = m0 + wm * 32 + r + 8 * khalf;
            float va = (bsel == 0) ? c00[r] : c01[r];
            float vb = (bsel == 0) ? c10[r] : c11[r];
            if (ma < M)      C[(size_t)ma * Ncols + nn]        = apply_mode(va, bv, mode);
            if (ma + 16 < M) C[(size_t)(ma + 16) * Ncols + nn] = apply_mode(vb, bv, mode);
        }
    }
}

// ---------------------------------------------------------------------------
// SpMM scatter: out[row[e], :] += val[e] * H[col[e], :]
// ---------------------------------------------------------------------------
__global__ void spmm_kernel(const int* __restrict__ row, const int* __restrict__ col,
                            const float* __restrict__ val, const float* __restrict__ H,
                            float* __restrict__ out, int D, long long total) {
    long long gid = (long long)blockIdx.x * blockDim.x + threadIdx.x;
    if (gid >= total) return;
    int e = (int)(gid / D);
    int j = (int)(gid - (long long)e * D);
    int r = row[e], c = col[e];
    float v = val[e] * H[(size_t)c * D + j];
    unsafeAtomicAdd(&out[(size_t)r * D + j], v);
}

__global__ void fill_zero_kernel(float* __restrict__ p, long long n) {
    long long i = (long long)blockIdx.x * blockDim.x + threadIdx.x;
    if (i < n) p[i] = 0.0f;
}

__global__ void bias_relu_kernel(float* __restrict__ x, const float* __restrict__ b,
                                 long long n, int cols) {
    long long i = (long long)blockIdx.x * blockDim.x + threadIdx.x;
    if (i >= n) return;
    int c = (int)(i % cols);
    x[i] = fmaxf(x[i] + b[c], 0.0f);
}

__global__ void bias_add_kernel(float* __restrict__ x, const float* __restrict__ b,
                                long long n, int cols) {
    long long i = (long long)blockIdx.x * blockDim.x + threadIdx.x;
    if (i >= n) return;
    int c = (int)(i % cols);
    x[i] = x[i] + b[c];
}

__global__ void add3_kernel(const float* __restrict__ a, const float* __restrict__ b,
                            const float* __restrict__ c, float* __restrict__ o, long long n) {
    long long i = (long long)blockIdx.x * blockDim.x + threadIdx.x;
    if (i < n) o[i] = a[i] + b[i] + c[i];
}

// Per-column mean/var over nrows (one block per column)
__global__ void bn_stats_kernel(const float* __restrict__ x, float* __restrict__ mu,
                                float* __restrict__ var, int nrows, int ncols) {
    __shared__ float ss[256];
    __shared__ float sq[256];
    int c = blockIdx.x;
    int tid = threadIdx.x;
    float s = 0.0f, q = 0.0f;
    for (int r = tid; r < nrows; r += 256) {
        float v = x[(size_t)r * ncols + c];
        s += v; q += v * v;
    }
    ss[tid] = s; sq[tid] = q;
    __syncthreads();
    for (int off = 128; off > 0; off >>= 1) {
        if (tid < off) { ss[tid] += ss[tid + off]; sq[tid] += sq[tid + off]; }
        __syncthreads();
    }
    if (tid == 0) {
        float m = ss[0] / (float)nrows;
        mu[c]  = m;
        var[c] = sq[0] / (float)nrows - m * m;
    }
}

__global__ void bn_apply_relu_kernel(const float* __restrict__ x, const float* __restrict__ g,
                                     const float* __restrict__ be, const float* __restrict__ mu,
                                     const float* __restrict__ var, float* __restrict__ y,
                                     long long n, int cols) {
    long long i = (long long)blockIdx.x * blockDim.x + threadIdx.x;
    if (i >= n) return;
    int c = (int)(i % cols);
    float v = g[c] * (x[i] - mu[c]) * rsqrtf(var[c] + EPS) + be[c];
    y[i] = fmaxf(v, 0.0f);
}

// Attention over z = [emb1, com/3, emb2, emb3]; writes fused embedding [N,64]
__global__ void attention_kernel(const float* __restrict__ e1, const float* __restrict__ cm,
                                 const float* __restrict__ e2, const float* __restrict__ e3,
                                 const float* __restrict__ w1, const float* __restrict__ b1,
                                 const float* __restrict__ w2, float* __restrict__ o, int n) {
    int i = blockIdx.x * blockDim.x + threadIdx.x;
    if (i >= n) return;
    const float* zp[4] = { e1 + (size_t)i * H2, cm + (size_t)i * H2,
                           e2 + (size_t)i * H2, e3 + (size_t)i * H2 };
    const float sc[4] = { 1.0f, 1.0f / 3.0f, 1.0f, 1.0f };
    float w[4];
    float mx = -1e30f;
    for (int j = 0; j < 4; ++j) {
        float acc = 0.0f;
        for (int k = 0; k < 16; ++k) {
            float t = b1[k];
            for (int d = 0; d < H2; ++d) t += zp[j][d] * sc[j] * w1[d * 16 + k];
            acc += tanhf(t) * w2[k];
        }
        w[j] = acc;
        mx = fmaxf(mx, acc);
    }
    float beta[4], s = 0.0f;
    for (int j = 0; j < 4; ++j) { beta[j] = expf(w[j] - mx); s += beta[j]; }
    for (int j = 0; j < 4; ++j) beta[j] /= s;
    for (int d = 0; d < H2; ++d) {
        float v = 0.0f;
        for (int j = 0; j < 4; ++j) v += beta[j] * zp[j][d] * sc[j];
        o[(size_t)i * H2 + d] = v;
    }
}

// Cluster head: y = softmax(z @ cl_w + cl_b), z is [N,64], cl_w [64,10]
__global__ void cluster_head_kernel(const float* __restrict__ z, const float* __restrict__ clw,
                                    const float* __restrict__ clb, float* __restrict__ y, int n) {
    int i = blockIdx.x * blockDim.x + threadIdx.x;
    if (i >= n) return;
    const float* zr = z + (size_t)i * H2;
    float lg[KCL];
    float mx = -1e30f;
    for (int k = 0; k < KCL; ++k) {
        float a = clb[k];
        for (int d = 0; d < H2; ++d) a += zr[d] * clw[d * KCL + k];
        lg[k] = a;
        mx = fmaxf(mx, a);
    }
    float s = 0.0f;
    for (int k = 0; k < KCL; ++k) { lg[k] = expf(lg[k] - mx); s += lg[k]; }
    for (int k = 0; k < KCL; ++k) y[(size_t)i * KCL + k] = lg[k] / s;
}

// ---------------------------------------------------------------------------
extern "C" void kernel_launch(void* const* d_in, const int* in_sizes, int n_in,
                              void* d_out, int out_size, void* d_ws, size_t ws_size,
                              hipStream_t stream) {
    (void)n_in; (void)out_size; (void)ws_size;
    // ---- inputs (setup_inputs dict order; params flattened jax-style: sorted keys)
    const float* x     = (const float*)d_in[0];
    const float* x_aug = (const float*)d_in[1];
    const int*   s_row = (const int*)d_in[2];
    const int*   s_col = (const int*)d_in[3];
    const float* s_val = (const float*)d_in[4];
    const int*   f_row = (const int*)d_in[5];
    const int*   f_col = (const int*)d_in[6];
    const float* f_val = (const float*)d_in[7];
    const int*   g_row = (const int*)d_in[8];
    const int*   g_col = (const int*)d_in[9];
    const float* g_val = (const float*)d_in[10];
    const int E = in_sizes[2];

    const float* agcn_b1 = (const float*)d_in[11];
    const float* agcn_b2 = (const float*)d_in[12];
    const float* agcn_w1 = (const float*)d_in[13];
    const float* agcn_w2 = (const float*)d_in[14];
    const float* att_b1  = (const float*)d_in[15];
    const float* att_w1  = (const float*)d_in[16];
    const float* att_w2  = (const float*)d_in[17];
    const float* cgcn_b1 = (const float*)d_in[18];
    const float* cgcn_b2 = (const float*)d_in[19];
    const float* cgcn_w1 = (const float*)d_in[20];
    const float* cgcn_w2 = (const float*)d_in[21];
    const float* cl_b    = (const float*)d_in[22];
    const float* cl_w    = (const float*)d_in[23];
    const float* cp_b    = (const float*)d_in[24];
    const float* cp_be   = (const float*)d_in[25];
    const float* cp_g    = (const float*)d_in[26];
    const float* cp_w    = (const float*)d_in[27];
    const float* dec_b   = (const float*)d_in[28];
    const float* dec_be  = (const float*)d_in[29];
    const float* dec_g   = (const float*)d_in[30];
    const float* dec_w   = (const float*)d_in[31];
    const float* disp_b  = (const float*)d_in[32];
    const float* disp_w  = (const float*)d_in[33];
    const float* fgcn_b1 = (const float*)d_in[34];
    const float* fgcn_b2 = (const float*)d_in[35];
    const float* fgcn_w1 = (const float*)d_in[36];
    const float* fgcn_w2 = (const float*)d_in[37];
    const float* mean_b  = (const float*)d_in[38];
    const float* mean_w  = (const float*)d_in[39];
    const float* mlp_b   = (const float*)d_in[40];
    const float* mlp_w   = (const float*)d_in[41];
    const float* pi_b    = (const float*)d_in[42];
    const float* pi_w    = (const float*)d_in[43];
    const float* sgcn_b1 = (const float*)d_in[44];
    const float* sgcn_b2 = (const float*)d_in[45];
    const float* sgcn_w1 = (const float*)d_in[46];
    const float* sgcn_w2 = (const float*)d_in[47];

    // ---- output layout (tuple concatenated flat)
    float* out = (float*)d_out;
    size_t o = 0;
    float* com1_o = out + o; o += (size_t)NN * H2;
    float* com2_o = out + o; o += (size_t)NN * H2;
    float* com3_o = out + o; o += (size_t)NN * H2;
    float* emb_o  = out + o; o += (size_t)NN * H2;
    float* pi_o   = out + o; o += (size_t)NN * NFEAT;
    float* disp_o = out + o; o += (size_t)NN * NFEAT;
    float* mean_o = out + o; o += (size_t)NN * NFEAT;
    float* y_o[5];
    float* p_o[5];
    for (int i = 0; i < 5; ++i) {
        y_o[i] = out + o; o += (size_t)NN * KCL;
        p_o[i] = out + o; o += (size_t)NN * H2;
    }

    // ---- workspace layout
    float* ws = (float*)d_ws;
    size_t w = 0;
    float* XW     = ws + w; w += (size_t)NN * H1;   // x @ w1 scratch
    float* Hbuf   = ws + w; w += (size_t)NN * H1;   // spmm1 / relu hidden
    float* DEC    = ws + w; w += (size_t)NN * H1;   // decoder pre-BN / h
    float* HW2    = ws + w; w += (size_t)NN * H2;   // h @ w2
    float* emb1   = ws + w; w += (size_t)NN * H2;
    float* emb2   = ws + w; w += (size_t)NN * H2;
    float* emb3   = ws + w; w += (size_t)NN * H2;
    float* comb   = ws + w; w += (size_t)NN * H2;
    float* embatt = ws + w; w += (size_t)NN * H2;
    float* clupre = ws + w; w += (size_t)NN * H2;
    float* mu_s   = ws + w; w += H1;
    float* var_s  = ws + w; w += H1;

    auto gemm = [&](const float* A, const float* B, const float* bias, float* C,
                    int M, int K, int Nc, int mode) {
        dim3 grid((Nc + BN - 1) / BN, (M + BM - 1) / BM);
        wmma_gemm_kernel<<<grid, 256, 0, stream>>>(A, B, bias, C, M, K, Nc, mode);
    };
    auto fill0 = [&](float* p, long long n) {
        fill_zero_kernel<<<(int)((n + 255) / 256), 256, 0, stream>>>(p, n);
    };
    auto spmm = [&](const int* er, const int* ec, const float* ev,
                    const float* Hm, float* op, int D) {
        long long tot = (long long)E * D;
        spmm_kernel<<<(int)((tot + 255) / 256), 256, 0, stream>>>(er, ec, ev, Hm, op, D, tot);
    };

    // GCN tail: given XWp = x@w1, do spmm1 -> relu(+b1) -> @w2 -> spmm2 -> +b2
    auto gcn_tail = [&](const float* XWp, const int* er, const int* ec, const float* ev,
                        const float* b1, const float* w2, const float* b2, float* outD) {
        long long n1 = (long long)NN * H1;
        long long n2 = (long long)NN * H2;
        fill0(Hbuf, n1);
        spmm(er, ec, ev, XWp, Hbuf, H1);
        bias_relu_kernel<<<(int)((n1 + 255) / 256), 256, 0, stream>>>(Hbuf, b1, n1, H1);
        gemm(Hbuf, w2, nullptr, HW2, NN, H1, H2, ACT_NONE);
        fill0(outD, n2);
        spmm(er, ec, ev, HW2, outD, H2);
        bias_add_kernel<<<(int)((n2 + 255) / 256), 256, 0, stream>>>(outD, b2, n2, H2);
    };

    // ---- six GCN passes (XW buffer reused; cgcn XW shared by com1/com2)
    gemm(x, sgcn_w1, nullptr, XW, NN, NFEAT, H1, ACT_NONE);
    gcn_tail(XW, s_row, s_col, s_val, sgcn_b1, sgcn_w2, sgcn_b2, emb1);

    gemm(x, cgcn_w1, nullptr, XW, NN, NFEAT, H1, ACT_NONE);
    gcn_tail(XW, s_row, s_col, s_val, cgcn_b1, cgcn_w2, cgcn_b2, com1_o);
    gcn_tail(XW, f_row, f_col, f_val, cgcn_b1, cgcn_w2, cgcn_b2, com2_o);

    gemm(x, fgcn_w1, nullptr, XW, NN, NFEAT, H1, ACT_NONE);
    gcn_tail(XW, f_row, f_col, f_val, fgcn_b1, fgcn_w2, fgcn_b2, emb2);

    gemm(x_aug, agcn_w1, nullptr, XW, NN, NFEAT, H1, ACT_NONE);
    gcn_tail(XW, g_row, g_col, g_val, agcn_b1, agcn_w2, agcn_b2, emb3);

    gemm(x_aug, cgcn_w1, nullptr, XW, NN, NFEAT, H1, ACT_NONE);
    gcn_tail(XW, g_row, g_col, g_val, cgcn_b1, cgcn_w2, cgcn_b2, com3_o);

    // ---- com, attention fusion, emb
    long long nc = (long long)NN * H2;
    add3_kernel<<<(int)((nc + 255) / 256), 256, 0, stream>>>(emb1, emb2, emb3, comb, nc);
    attention_kernel<<<(NN + 127) / 128, 128, 0, stream>>>(
        emb1, comb, emb2, emb3, att_w1, att_b1, att_w2, embatt, NN);
    gemm(embatt, mlp_w, mlp_b, emb_o, NN, H2, H2, ACT_BIAS);

    // ---- decoder: BN + relu, then fused heads straight into d_out
    long long nd = (long long)NN * H1;
    gemm(emb_o, dec_w, dec_b, DEC, NN, H2, H1, ACT_BIAS);
    bn_stats_kernel<<<H1, 256, 0, stream>>>(DEC, mu_s, var_s, NN, H1);
    bn_apply_relu_kernel<<<(int)((nd + 255) / 256), 256, 0, stream>>>(
        DEC, dec_g, dec_be, mu_s, var_s, DEC, nd, H1);
    gemm(DEC, pi_w, pi_b, pi_o, NN, H1, NFEAT, ACT_SIGMOID);
    gemm(DEC, disp_w, disp_b, disp_o, NN, H1, NFEAT, ACT_SOFTPLUS_CLIP);
    gemm(DEC, mean_w, mean_b, mean_o, NN, H1, NFEAT, ACT_EXP_CLIP);

    // ---- five cluster heads: (com1, com2, com3, com, emb)
    const float* cin[5] = { com1_o, com2_o, com3_o, comb, emb_o };
    for (int i = 0; i < 5; ++i) {
        gemm(cin[i], cp_w, cp_b, clupre, NN, H2, H2, ACT_BIAS);
        bn_stats_kernel<<<H2, 256, 0, stream>>>(clupre, mu_s, var_s, NN, H2);
        bn_apply_relu_kernel<<<(int)((nc + 255) / 256), 256, 0, stream>>>(
            clupre, cp_g, cp_be, mu_s, var_s, p_o[i], nc, H2);
        cluster_head_kernel<<<(NN + 127) / 128, 128, 0, stream>>>(
            p_o[i], cl_w, cl_b, y_o[i], NN);
    }
}
